// LinearBase_71743133713048
// MI455X (gfx1250) — matrix operand order repaired and checked by reading, not compiled, not run on hardware
//
#include <hip/hip_runtime.h>

typedef __attribute__((ext_vector_type(16))) _Float16 v16h;
typedef __attribute__((ext_vector_type(8)))  float    v8f;
typedef __attribute__((ext_vector_type(4)))  float    f32x4;
typedef __attribute__((ext_vector_type(4)))  int      i32x4;

constexpr int Mdim  = 8;
constexpr int Kdim  = 8192;
constexpr int Ndim  = 28672;
constexpr int WAVES = 8;                 // waves per block
constexpr int KSLICE = Kdim / WAVES;     // 1024 K per wave
constexpr int NTILE  = 16;               // output columns per block

// y[m,n] = scales[n] * sum_k x[m,k] * q[n,k]
// One 16-wide N tile per block; 8 waves split K; each wave accumulates a
// 16x16 f32 tile via v_wmma_f32_16x16x32_f16; LDS reduction (serial 8-way sum per slot) at the end.
//
// Bandwidth analysis (MI455X, 23.3 TB/s): qweight = 940 MB read once (NT
// hints, streamed); x = 256 KB, re-read 1792x but L2-resident; kernel floor
// ~40 us. Required compute 4 FLOP/B * 23.3 TB/s ~ 93 TFLOP/s, well inside
// the f16 WMMA ceiling. Weights are int8-range -> exact in f16.
__global__ __launch_bounds__(256) void w8a16_wmma_kernel(
    const float* __restrict__ x,      // [8, 8192] fp32
    const int*   __restrict__ qw,     // [28672, 8192] int32 (int8 range)
    const float* __restrict__ scales, // [28672] fp32
    float*       __restrict__ y)      // [8, 28672] fp32
{
    __shared__ float red[WAVES * 256];   // 8 KB

    const int tid  = threadIdx.x;
    const int wave = tid >> 5;
    const int lane = tid & 31;
    const int hsel = lane >> 4;          // 0: lanes 0-15, 1: lanes 16-31
    const int l16  = lane & 15;

    const int  ntile = blockIdx.x * NTILE;
    const int  n     = ntile + l16;                  // B column owned by lane
    const long nrow  = (long)n * Kdim;

    // A row owned by this lane. Rows 8..15 of the A tile are dead weight
    // (M=8 < 16): clamp them onto rows 0..7. Their products land only in
    // C rows 8..15, which are discarded at writeback -> no predication, no
    // zero-fill, and the duplicate loads are exact L0/L2 hits.
    const int   m    = l16 & (Mdim - 1);
    const float* xrow = x + (long)m * Kdim;

    const int kw = wave * KSLICE;

    v8f c = {};

    #pragma unroll 2
    for (int kk = 0; kk < KSLICE; kk += 32) {
        const int kbase = kw + kk;

        // ---------- B: 16 contiguous int32 weights -> 16 f16 (exact) -------
        // lanes 0-15: K = kbase+0..15 ; lanes 16-31: K = kbase+16..31
        // One 64B cacheline per lane; non-temporal (read-once stream).
        const int kb = kbase + hsel * 16;
        const i32x4* qp = (const i32x4*)(qw + nrow + kb);   // 64B aligned
        const i32x4 q0 = __builtin_nontemporal_load(qp + 0);
        const i32x4 q1 = __builtin_nontemporal_load(qp + 1);
        const i32x4 q2 = __builtin_nontemporal_load(qp + 2);
        const i32x4 q3 = __builtin_nontemporal_load(qp + 3);
        // Pull the next iteration's cacheline toward us. At worst this runs
        // a few bytes past the 940MB array (same page); speculative
        // prefetches that miss translation are silently dropped.
        __builtin_prefetch(qw + nrow + kb + 32, 0, 1);

        v16h b;
        #pragma unroll
        for (int j = 0; j < 4; ++j) {
            b[j]      = (_Float16)(float)q0[j];
            b[4 + j]  = (_Float16)(float)q1[j];
            b[8 + j]  = (_Float16)(float)q2[j];
            b[12 + j] = (_Float16)(float)q3[j];
        }

        // ---------- A: activations fp32 -> f16 (L2-resident) ---------------
        // lanes 0-15: K = kbase+0..15 ; lanes 16-31: K = kbase+16..31
        const int ab = kbase + hsel * 16;
        const f32x4 x0 = *(const f32x4*)(xrow + ab);
        const f32x4 x1 = *(const f32x4*)(xrow + ab + 4);
        const f32x4 x2 = *(const f32x4*)(xrow + ab + 8);
        const f32x4 x3 = *(const f32x4*)(xrow + ab + 12);

        v16h a;
        #pragma unroll
        for (int j = 0; j < 4; ++j) {
            a[j]      = (_Float16)x0[j];
            a[4 + j]  = (_Float16)x1[j];
            a[8 + j]  = (_Float16)x2[j];
            a[12 + j] = (_Float16)x3[j];
        }

        // ---------- D = A x B + C (fp32 accumulate) ------------------------
        c = __builtin_amdgcn_wmma_f32_16x16x32_f16(
                /*neg_a=*/false, a, /*neg_b=*/false, b,
                /*c_mod=*/(short)0, c, /*reuse_a=*/false, /*reuse_b=*/false);
    }

    // ---------- cross-wave K reduction via LDS -----------------------------
    #pragma unroll
    for (int r = 0; r < 8; ++r)
        red[wave * 256 + r * 32 + lane] = c[r];
    __syncthreads();

    float s = 0.0f;
    #pragma unroll
    for (int w = 0; w < WAVES; ++w)
        s += red[w * 256 + tid];

    // slot tid -> (VGPR r = tid>>5, lane' = tid&31); M = r + 8*(lane'>=16),
    // N = lane'&15. Valid output rows are M<8 <=> lane'<16.
    const int r  = tid >> 5;
    const int ln = tid & 31;
    if (ln < 16) {
        const int col = ntile + ln;
        y[(long)r * Ndim + col] = s * scales[col];
    }
}

extern "C" void kernel_launch(void* const* d_in, const int* in_sizes, int n_in,
                              void* d_out, int out_size, void* d_ws, size_t ws_size,
                              hipStream_t stream) {
    (void)in_sizes; (void)n_in; (void)out_size; (void)d_ws; (void)ws_size;
    const float* x      = (const float*)d_in[0];
    const int*   qw     = (const int*)d_in[1];
    const float* scales = (const float*)d_in[2];
    float*       y      = (float*)d_out;

    dim3 grid(Ndim / NTILE);   // 1792 blocks
    dim3 block(256);           // 8 waves
    hipLaunchKernelGGL(w8a16_wmma_kernel, grid, block, 0, stream,
                       x, qw, scales, y);
}
